// ProbSparseAttention_56573309223405
// MI455X (gfx1250) — matrix operand order repaired and checked by reading, not compiled, not run on hardware
//
#include <hip/hip_runtime.h>
#include <hip/hip_bf16.h>

// Problem constants (from reference): B,L,S,H,E,D
#define PB 2
#define PL 2048
#define PS 2048
#define PH 8
#define PE 64
#define PD 64

typedef _Float16 half16 __attribute__((ext_vector_type(16)));
typedef float    v8f    __attribute__((ext_vector_type(8)));

union AFrag {
    half16       v;
    unsigned int u[8];
};

__device__ __forceinline__ unsigned short f2h(float x) {
    _Float16 h = (_Float16)x;
    return __builtin_bit_cast(unsigned short, h);
}
__device__ __forceinline__ unsigned int pack2(float a, float b) {
    return (unsigned int)f2h(a) | ((unsigned int)f2h(b) << 16);
}
// butterfly reductions across a 16-lane half (used once per pass)
__device__ __forceinline__ float half_max(float v) {
    #pragma unroll
    for (int m = 1; m < 16; m <<= 1) v = fmaxf(v, __shfl_xor(v, m, 32));
    return v;
}
__device__ __forceinline__ float half_sum(float v) {
    #pragma unroll
    for (int m = 1; m < 16; m <<= 1) v += __shfl_xor(v, m, 32);
    return v;
}
// CDNA5 async copy: 32B global -> LDS per thread (offset applies to BOTH sides)
__device__ __forceinline__ void async_copy32(unsigned int lds_off, const void* g) {
    unsigned long long ga = (unsigned long long)(uintptr_t)g;
    asm volatile("global_load_async_to_lds_b128 %0, %1, off\n\t"
                 "global_load_async_to_lds_b128 %0, %1, off offset:16"
                 :: "v"(lds_off), "v"(ga) : "memory");
}
// async loads retire in order -> waiting for <=N keeps newest N in flight
template<int N>
__device__ __forceinline__ void wait_async_le() {
    asm volatile("s_wait_asynccnt %0" :: "i"(N) : "memory");
}

// ---------------------------------------------------------------------------
// Kernel 0a: K (f32, [B,S,H,E]) -> Kh (f16, [B,H,S,E])  (head-major rows)
// ---------------------------------------------------------------------------
__global__ __launch_bounds__(256)
void probsparse_convk_kernel(const float* __restrict__ Kg,
                             unsigned short* __restrict__ Kh) {
    size_t o = ((size_t)blockIdx.x * 256 + threadIdx.x) * 8;
    int    e    = (int)(o & (PE - 1));
    size_t rest = o / PE;
    int    s    = (int)(rest & (PS - 1));
    size_t bh   = rest / PS;
    int    h    = (int)(bh & (PH - 1));
    int    b    = (int)(bh / PH);
    const float* src = Kg + (((size_t)b * PS + s) * PH + h) * PE + e;
    unsigned int* dst = (unsigned int*)(Kh + o);
    #pragma unroll
    for (int j = 0; j < 4; ++j) dst[j] = pack2(src[2 * j], src[2 * j + 1]);
}

// ---------------------------------------------------------------------------
// Kernel 0b: V (f32, [B,S,H,D]) -> VTh (f16, [B,H,D,S])  (transposed)
// ---------------------------------------------------------------------------
__global__ __launch_bounds__(256)
void probsparse_convv_kernel(const float* __restrict__ Vg,
                             unsigned short* __restrict__ VTh) {
    const int stile = blockIdx.x % (PS / 64);
    const int bh    = blockIdx.x / (PS / 64);
    const int b = bh / PH, h = bh % PH;

    __shared__ unsigned short tile[64][72];   // padded rows

    const int row = threadIdx.x >> 2;
    const int c   = (threadIdx.x & 3) * 16;

    const float* src = Vg + (((size_t)b * PS + stile * 64 + row) * PH + h) * PD + c;
    #pragma unroll
    for (int j = 0; j < 16; ++j) tile[row][c + j] = f2h(src[j]);
    __syncthreads();

    unsigned short* dst = VTh + ((size_t)bh * PD + row) * PS + stile * 64 + c;
    #pragma unroll
    for (int j = 0; j < 16; ++j) dst[j] = tile[c + j][row];
}

// ---------------------------------------------------------------------------
// Kernel 1: two-pass flash attention (unpermuted) + M_sp statistics.
// grid = B*H*(L/64), block = 128 (4 waves, 16 queries per wave)
// Double-buffered GLOBAL_LOAD_ASYNC_TO_LDS staging overlapped with WMMA.
// ---------------------------------------------------------------------------
__global__ __launch_bounds__(128)
void probsparse_flash_kernel(const float* __restrict__ Q,
                             const unsigned short* __restrict__ Kh,   // [B,H,S,E] f16
                             const unsigned short* __restrict__ VTh,  // [B,H,D,S] f16
                             float* __restrict__ Vtmp,   // [B,H,L,D]
                             float* __restrict__ Msp) {  // [B,H,L]
    const int lane = threadIdx.x & 31;
    const int wave = threadIdx.x >> 5;
    const int n16  = lane & 15;
    const int half = lane >> 4;

    const int qtiles = PL / 64;
    const int qt = blockIdx.x % qtiles;
    const int h  = (blockIdx.x / qtiles) % PH;
    const int b  =  blockIdx.x / (qtiles * PH);

    __shared__ unsigned short sK [2][32 * 64];   // K chunks  (2 x 4KB)
    __shared__ unsigned short sVT[2][64 * 32];   // V^T chunks (2 x 4KB)
    __shared__ unsigned short sP [4][16 * 32];   // per-wave P bounce (4KB)

    const unsigned int sK_off  = (unsigned int)(uintptr_t)&sK[0][0];
    const unsigned int sVT_off = (unsigned int)(uintptr_t)&sVT[0][0];

    // ---- load Q A-fragments once (contiguous k = 16*half + s, 2 k-steps of 32) ----
    const int qrow = qt * 64 + wave * 16 + n16;
    const float* qptr = Q + (((size_t)b * PL + qrow) * PH + h) * PE;
    AFrag aq[2];
    #pragma unroll
    for (int ks = 0; ks < 2; ++ks) {
        #pragma unroll
        for (int v = 0; v < 8; ++v) {
            int kk = ks * 32 + 2 * v + 16 * half;
            aq[ks].u[v] = pack2(qptr[kk], qptr[kk + 1]);
        }
    }

    const float inv_sqrt_e = 0.125f;
    const v8f vzero = {0.f, 0.f, 0.f, 0.f, 0.f, 0.f, 0.f, 0.f};

    const unsigned short* khp  = Kh  + ((size_t)b * PH + h) * PS * PE;
    const unsigned short* vthp = VTh + ((size_t)b * PH + h) * PD * PS;

    const int vrow = threadIdx.x >> 1;          // d row 0..63 (V staging)
    const int vseg = (threadIdx.x & 1) * 32;    // byte segment within 64B row

    const int NCH = PS / 32;

    // =======================  PASS A: raw max / raw sum  ====================
    float mpart[8], spart[8];
    #pragma unroll
    for (int r = 0; r < 8; ++r) { mpart[r] = -3.0e38f; spart[r] = 0.f; }

    // prologue: chunk 0 -> buffer 0 (2 async instrs per thread per chunk)
    async_copy32(sK_off + threadIdx.x * 32, (const char*)khp + threadIdx.x * 32);

    for (int it = 0; it < NCH; ++it) {
        __syncthreads();   // everyone done computing from the buffer we overwrite
        if (it + 1 < NCH) {
            const char* kc = (const char*)(khp + (size_t)(it + 1) * 32 * PE)
                             + threadIdx.x * 32;
            async_copy32(sK_off + ((it + 1) & 1) * 4096 + threadIdx.x * 32, kc);
            __builtin_prefetch(kc + 4096, 0, 3);   // chunk it+2 -> L2
            wait_async_le<2>();                    // chunk it retired, it+1 in flight
        } else {
            wait_async_le<0>();
        }
        __syncthreads();

        const unsigned int* kp = (const unsigned int*)sK[it & 1];
        v8f c0 = vzero, c1 = vzero;
        #pragma unroll
        for (int ks = 0; ks < 2; ++ks) {
            AFrag bk;
            #pragma unroll
            for (int v = 0; v < 8; ++v)
                bk.u[v] = kp[(0 + n16) * 32 + ks * 16 + 8 * half + v];
            c0 = __builtin_amdgcn_wmma_f32_16x16x32_f16(false, aq[ks].v, false, bk.v,
                                                        (short)0, c0, false, false);
            #pragma unroll
            for (int v = 0; v < 8; ++v)
                bk.u[v] = kp[(16 + n16) * 32 + ks * 16 + 8 * half + v];
            c1 = __builtin_amdgcn_wmma_f32_16x16x32_f16(false, aq[ks].v, false, bk.v,
                                                        (short)0, c1, false, false);
        }
        #pragma unroll
        for (int r = 0; r < 8; ++r) {
            mpart[r]  = fmaxf(mpart[r], fmaxf(c0[r], c1[r]));
            spart[r] += c0[r] + c1[r];
        }
    }

    float mrow[8];
    #pragma unroll
    for (int r = 0; r < 8; ++r) {
        mrow[r] = half_max(mpart[r]);
        float srow = half_sum(spart[r]);
        if (n16 == 0) {
            int q = qt * 64 + wave * 16 + r + 8 * half;
            Msp[((size_t)b * PH + h) * PL + q] = mrow[r] - srow * (1.0f / (float)PS);
        }
    }

    // =================  PASS B: exp / PV accumulate / exp-sum  ==============
    float lpart[8];
    v8f o[4];
    #pragma unroll
    for (int r = 0; r < 8; ++r) lpart[r] = 0.f;
    #pragma unroll
    for (int f = 0; f < 4; ++f) o[f] = vzero;

    __syncthreads();   // pass A reads of buffers complete before re-staging
    // prologue: chunk 0 -> buffer 0 (4 async instrs per thread per chunk)
    async_copy32(sK_off + threadIdx.x * 32, (const char*)khp + threadIdx.x * 32);
    async_copy32(sVT_off + vrow * 64 + vseg,
                 (const char*)(vthp + (size_t)vrow * PS) + vseg);

    for (int it = 0; it < NCH; ++it) {
        __syncthreads();
        if (it + 1 < NCH) {
            int s0n = (it + 1) * 32;
            int buf = (it + 1) & 1;
            const char* kc = (const char*)(khp + (size_t)s0n * PE) + threadIdx.x * 32;
            async_copy32(sK_off + buf * 4096 + threadIdx.x * 32, kc);
            const char* vc = (const char*)(vthp + (size_t)vrow * PS + s0n) + vseg;
            async_copy32(sVT_off + buf * 4096 + vrow * 64 + vseg, vc);
            __builtin_prefetch(vc + 64, 0, 3);     // chunk it+2 V rows -> L2
            wait_async_le<4>();                    // chunk it retired, it+1 in flight
        } else {
            wait_async_le<0>();
        }
        __syncthreads();

        const unsigned int* kp  = (const unsigned int*)sK [it & 1];
        const unsigned int* vtp = (const unsigned int*)sVT[it & 1];

        v8f c0 = vzero, c1 = vzero;
        #pragma unroll
        for (int ks = 0; ks < 2; ++ks) {
            AFrag bk;
            #pragma unroll
            for (int v = 0; v < 8; ++v)
                bk.u[v] = kp[(0 + n16) * 32 + ks * 16 + 8 * half + v];
            c0 = __builtin_amdgcn_wmma_f32_16x16x32_f16(false, aq[ks].v, false, bk.v,
                                                        (short)0, c0, false, false);
            #pragma unroll
            for (int v = 0; v < 8; ++v)
                bk.u[v] = kp[(16 + n16) * 32 + ks * 16 + 8 * half + v];
            c1 = __builtin_amdgcn_wmma_f32_16x16x32_f16(false, aq[ks].v, false, bk.v,
                                                        (short)0, c1, false, false);
        }

        float p0[8], p1[8];
        #pragma unroll
        for (int r = 0; r < 8; ++r) {
            p0[r] = __expf((c0[r] - mrow[r]) * inv_sqrt_e);
            p1[r] = __expf((c1[r] - mrow[r]) * inv_sqrt_e);
            lpart[r] += p0[r] + p1[r];
        }

        // C-layout P -> A-layout via per-wave LDS bounce
        unsigned short* pw = sP[wave];
        #pragma unroll
        for (int r = 0; r < 8; ++r) {
            int m = r + 8 * half;
            pw[m * 32 + n16]      = f2h(p0[r]);
            pw[m * 32 + 16 + n16] = f2h(p1[r]);
        }
        asm volatile("s_wait_dscnt 0" ::: "memory");
        AFrag pa;
        const unsigned int* pp = (const unsigned int*)pw;
        #pragma unroll
        for (int v = 0; v < 8; ++v) {
            int pair = v + 8 * half;
            pa.u[v] = pp[n16 * 16 + pair];
        }

        #pragma unroll
        for (int f = 0; f < 4; ++f) {
            AFrag bv;
            #pragma unroll
            for (int v = 0; v < 8; ++v)
                bv.u[v] = vtp[(f * 16 + n16) * 16 + 8 * half + v];
            o[f] = __builtin_amdgcn_wmma_f32_16x16x32_f16(false, pa.v, false, bv.v,
                                                          (short)0, o[f], false, false);
        }
    }

    float lrow[8];
    #pragma unroll
    for (int r = 0; r < 8; ++r) {
        float l = half_sum(lpart[r]);
        lrow[r] = (l > 0.f) ? (1.0f / l) : 0.f;
    }
    #pragma unroll
    for (int f = 0; f < 4; ++f) {
        #pragma unroll
        for (int r = 0; r < 8; ++r) {
            int q = qt * 64 + wave * 16 + r + 8 * half;
            Vtmp[(((size_t)b * PH + h) * PL + q) * PD + f * 16 + n16] = o[f][r] * lrow[r];
        }
    }
}

// ---------------------------------------------------------------------------
// Kernel 2: per-(b,h) descending stable sort of M_sp (k == L) + row gather.
// ---------------------------------------------------------------------------
__global__ __launch_bounds__(256)
void probsparse_sort_gather_kernel(const float* __restrict__ Msp,
                                   const float* __restrict__ Vtmp,
                                   float* __restrict__ out) {
    const int bh = blockIdx.x;
    const int b  = bh / PH;
    const int h  = bh % PH;

    __shared__ float skey[PL];
    __shared__ int   sidx[PL];

    const float* ms = Msp + (size_t)bh * PL;
    for (int i = threadIdx.x; i < PL; i += 256) { skey[i] = ms[i]; sidx[i] = i; }
    __syncthreads();

    for (int k = 2; k <= PL; k <<= 1) {
        for (int j = k >> 1; j > 0; j >>= 1) {
            for (int t = threadIdx.x; t < PL / 2; t += 256) {
                int i = ((t & ~(j - 1)) << 1) | (t & (j - 1));
                int l = i | j;
                float ka = skey[i], kb = skey[l];
                int   ia = sidx[i], ib = sidx[l];
                bool aBeforeB = (ka > kb) || (ka == kb && ia < ib);
                bool wantA    = ((i & k) == 0);
                if (aBeforeB != wantA) {
                    skey[i] = kb; skey[l] = ka;
                    sidx[i] = ib; sidx[l] = ia;
                }
            }
            __syncthreads();
        }
    }

    const float* vt = Vtmp + (size_t)bh * PL * PD;
    float* ob = out + (size_t)b * PL * PH * PD + (size_t)h * PD;
    for (int e = threadIdx.x; e < PL * PD; e += 256) {
        int i = e >> 6;
        int d = e & 63;
        ob[(size_t)i * PH * PD + d] = vt[(size_t)sidx[i] * PD + d];
    }
}

// ---------------------------------------------------------------------------
extern "C" void kernel_launch(void* const* d_in, const int* in_sizes, int n_in,
                              void* d_out, int out_size, void* d_ws, size_t ws_size,
                              hipStream_t stream) {
    const float* Q  = (const float*)d_in[0];
    const float* Kg = (const float*)d_in[1];
    const float* Vg = (const float*)d_in[2];
    float* out = (float*)d_out;

    char* ws = (char*)d_ws;
    float*          Vtmp = (float*)ws;                       // B*H*L*D f32 (8 MB)
    ws += (size_t)PB * PH * PL * PD * sizeof(float);
    float*          Msp  = (float*)ws;                       // B*H*L f32 (128 KB)
    ws += (size_t)PB * PH * PL * sizeof(float);
    unsigned short* Kh   = (unsigned short*)ws;              // B*H*S*E f16 (4 MB)
    ws += (size_t)PB * PH * PS * PE * sizeof(unsigned short);
    unsigned short* VTh  = (unsigned short*)ws;              // B*H*D*S f16 (4 MB)

    {   // K convert: 8 elems/thread
        size_t total = (size_t)PB * PH * PS * PE / 8;
        probsparse_convk_kernel<<<dim3((unsigned)(total / 256)), 256, 0, stream>>>(Kg, Kh);
    }
    probsparse_convv_kernel<<<dim3(PB * PH * (PS / 64)), 256, 0, stream>>>(Vg, VTh);

    probsparse_flash_kernel<<<dim3(PB * PH * (PL / 64)), 128, 0, stream>>>(Q, Kh, VTh,
                                                                           Vtmp, Msp);
    probsparse_sort_gather_kernel<<<dim3(PB * PH), 256, 0, stream>>>(Msp, Vtmp, out);
}
